// EdgeBlock_9148280340719
// MI455X (gfx1250) — matrix operand order repaired and checked by reading, not compiled, not run on hardware
//
#include <hip/hip_runtime.h>

typedef __attribute__((ext_vector_type(16))) __bf16    v16bf;
typedef __attribute__((ext_vector_type(8)))  float     v8f;
typedef __attribute__((ext_vector_type(4)))  unsigned  v4u;
typedef __attribute__((ext_vector_type(8)))  unsigned  v8u;
typedef unsigned short ushort_t;

#define NODE_DIM 128
#define EDGE_DIM 64
#define OUT_DIM  128
#define NTILES   8        // OUT_DIM / 16
#define EB       32       // edges per block (2 M-tiles of 16)
#define SA       136      // bf16 LDS row stride for 128-wide rows (272B, 16B aligned)
#define SE       72       // bf16 LDS row stride for 64-wide rows  (144B, 16B aligned)
#define STRF     132      // f32 LDS row stride for the pre-LN output
#define LN_EPS   1e-5f

// ---------------------------------------------------------------------------
// bf16 hi/lo split helpers: v = hi + lo with |err| ~ 2^-17 |v|
// ---------------------------------------------------------------------------
__device__ __forceinline__ void split_bf(float v, ushort_t &h, ushort_t &l) {
  unsigned u = __builtin_bit_cast(unsigned, v);
  h = (ushort_t)(u >> 16);
  float hf = __builtin_bit_cast(float, u & 0xFFFF0000u);
  float r  = v - hf;
  l = (ushort_t)(__builtin_bit_cast(unsigned, r) >> 16);
}

// ---------------------------------------------------------------------------
// bf16 WMMA: D(16x16,f32) += A(16x32,bf16) * B(32x16,bf16)   [probe-confirmed]
// ---------------------------------------------------------------------------
__device__ __forceinline__ v8f wmma_bf(v16bf a, v16bf b, v8f c) {
  return __builtin_amdgcn_wmma_f32_16x16x32_bf16(false, a, false, b, (short)0,
                                                 c, false, false);
}

__device__ __forceinline__ v8f splat8(float v) {
  v8f r;
#pragma unroll
  for (int i = 0; i < 8; ++i) r[i] = v;
  return r;
}

__device__ __forceinline__ void relu8(v8f &a) {
#pragma unroll
  for (int i = 0; i < 8; ++i) a[i] = a[i] > 0.f ? a[i] : 0.f;
}

// A fragment: per lane two contiguous 16B chunks, 16 elements apart (K and K+16)
__device__ __forceinline__ v16bf ldfrag_a(const ushort_t *p) {
  v8u q;
  *(v4u *)&q       = *(const v4u *)(p);
  *((v4u *)&q + 1) = *(const v4u *)(p + 16);
  return __builtin_bit_cast(v16bf, q);
}
// B fragment: per lane 32 contiguous bytes
__device__ __forceinline__ v16bf ldfrag_b(const ushort_t *p) {
  v8u q;
  *(v4u *)&q       = *(const v4u *)(p);
  *((v4u *)&q + 1) = *(const v4u *)(p + 8);
  return __builtin_bit_cast(v16bf, q);
}

// Accumulate 2 M-tiles against `kchunks` K=32 chunks using the 3-term bf16
// split product: hi*hi + hi*lo + lo*hi (fp32 accumulate).
__device__ __forceinline__ void mm_bf16(v8f &acc0, v8f &acc1,
                                        const ushort_t *AH, const ushort_t *AL,
                                        int sA,
                                        const ushort_t *__restrict__ BB,
                                        int lane, int kchunks) {
  const int m     = lane & 15;
  const int khalf = (lane >> 4) << 3;   // lanes 16-31 start at K+8 / K+24
  const ushort_t *a0H = AH + m * sA + khalf;
  const ushort_t *a0L = AL + m * sA + khalf;
  const ushort_t *a1H = AH + (16 + m) * sA + khalf;
  const ushort_t *a1L = AL + (16 + m) * sA + khalf;
#pragma unroll
  for (int kc = 0; kc < kchunks; ++kc) {
    const ushort_t *bp = BB + (size_t)kc * 1024 + lane * 16;
    v16bf bh = ldfrag_b(bp);         // 2x global_load_b128, coalesced
    v16bf bl = ldfrag_b(bp + 512);
    const int ko = kc * 32;
    v16bf a0h = ldfrag_a(a0H + ko);  // 2x ds_load_b128 each
    v16bf a1h = ldfrag_a(a1H + ko);
    v16bf a0l = ldfrag_a(a0L + ko);
    v16bf a1l = ldfrag_a(a1L + ko);
    acc0 = wmma_bf(a0h, bh, acc0);
    acc1 = wmma_bf(a1h, bh, acc1);
    acc0 = wmma_bf(a0h, bl, acc0);
    acc1 = wmma_bf(a1h, bl, acc1);
    acc0 = wmma_bf(a0l, bh, acc0);
    acc1 = wmma_bf(a1l, bh, acc1);
  }
}

// Scatter a 16x16 f32 D tile into hi/lo bf16 LDS planes (relu'd activation).
__device__ __forceinline__ void store_tile_bf(ushort_t *H, ushort_t *L, v8f acc,
                                              int mt, int wave, int lane) {
  const int col   = wave * 16 + (lane & 15);
  const int rbase = mt * 16 + ((lane >> 4) << 3);
#pragma unroll
  for (int r = 0; r < 8; ++r) {
    ushort_t h, l;
    split_bf(acc[r], h, l);
    H[(rbase + r) * SA + col] = h;
    L[(rbase + r) * SA + col] = l;
  }
}
// Scatter a 16x16 f32 D tile into the f32 pre-LN buffer.
__device__ __forceinline__ void store_tile_f32(float *buf, v8f acc, int mt,
                                               int wave, int lane) {
  const int col   = wave * 16 + (lane & 15);
  const int rbase = mt * 16 + ((lane >> 4) << 3);
#pragma unroll
  for (int r = 0; r < 8; ++r) buf[(rbase + r) * STRF + col] = acc[r];
}

// ---------------------------------------------------------------------------
// Repack W[OUT_DIM, K] (torch Linear, row-major) into bf16 hi/lo WMMA-B
// fragments. Fragment (ntile,kchunk): lane l holds W_bf[nt*16 + l%16]
// [kc*32 + (l/16)*8 + {0..7, 16..23}]; hi frag then lo frag, 1024 ushorts per pair.
// ---------------------------------------------------------------------------
__global__ void pack_wt_bf(const float *__restrict__ W,
                           ushort_t *__restrict__ dst, int K) {
  const int kchunks = K >> 5;
  const int total   = NTILES * kchunks * 32;
  const int tid = blockIdx.x * blockDim.x + threadIdx.x;
  if (tid >= total) return;
  const int lane = tid & 31;
  const int f    = tid >> 5;
  const int kc   = f % kchunks;
  const int nt   = f / kchunks;
  const int n    = nt * 16 + (lane & 15);
  const int kb   = kc * 32 + ((lane >> 4) << 3);
  const float *w = W + (size_t)n * K + kb;
  ushort_t *hq = dst + (size_t)f * 1024 + lane * 16;
  ushort_t *lq = hq + 512;
#pragma unroll
  for (int j = 0; j < 16; ++j) {
    ushort_t h, l;
    split_bf(w[j + (j & 8)], h, l);
    hq[j] = h;
    lq[j] = l;
  }
}

// ---------------------------------------------------------------------------
// Main fused kernel: 32 edges/block, both directions, averaged + skip + LN.
// ---------------------------------------------------------------------------
__global__ __launch_bounds__(256, 2)
void edge_block_wmma(const float *__restrict__ x,
                     const long long *__restrict__ eidx,
                     const float *__restrict__ ea,
                     const float *__restrict__ b_edge,
                     const float *__restrict__ b_node,
                     const float *__restrict__ b_comb,
                     const float *__restrict__ gamma,
                     const float *__restrict__ beta,
                     const ushort_t *__restrict__ BN,
                     const ushort_t *__restrict__ BC,
                     const ushort_t *__restrict__ BE,
                     const ushort_t *__restrict__ BS,
                     float *__restrict__ out, int E) {
  __shared__ __attribute__((aligned(16))) ushort_t sXsH[EB * SA], sXsL[EB * SA];
  __shared__ __attribute__((aligned(16))) ushort_t sXdH[EB * SA], sXdL[EB * SA];
  __shared__ __attribute__((aligned(16))) ushort_t sPNH[EB * SA], sPNL[EB * SA];
  __shared__ __attribute__((aligned(16))) ushort_t sPEH[EB * SA], sPEL[EB * SA];
  __shared__ __attribute__((aligned(16))) ushort_t sEAH[EB * SE], sEAL[EB * SE];
  __shared__ __attribute__((aligned(16))) float    sO[EB * STRF];
  __shared__ float sRs[256], sRss[256];
  __shared__ float sMU[EB], sRSD[EB];
  __shared__ int   sSrc[EB], sDst[EB];

  const int t   = threadIdx.x;
  const int eb0 = blockIdx.x * EB;

  if (eb0 + EB < E)  // warm L2/WGP$ for the next block's edge_attr tile
    __builtin_prefetch(ea + (size_t)(eb0 + EB) * EDGE_DIM, 0, 3);

  if (t < EB) {
    int e = eb0 + t;
    if (e > E - 1) e = E - 1;
    sSrc[t] = (int)eidx[e];               // edge_index[0][e]
    sDst[t] = (int)eidx[(size_t)E + e];   // edge_index[1][e]
  }
  __syncthreads();

  // Gather x[src]/x[dst] rows, split f32 -> bf16 hi/lo planes on the fly.
  for (int i = t; i < 2 * EB * 32; i += 256) {   // 64 rows x 32 float4
    const int r  = i >> 5;
    const int c4 = i & 31;
    const int node = (r < EB) ? sSrc[r] : sDst[r - EB];
    float4 v = ((const float4 *)(x + (size_t)node * NODE_DIM))[c4];
    ushort_t h0, l0, h1, l1, h2, l2, h3, l3;
    split_bf(v.x, h0, l0); split_bf(v.y, h1, l1);
    split_bf(v.z, h2, l2); split_bf(v.w, h3, l3);
    uint2 hp, lp;
    hp.x = (unsigned)h0 | ((unsigned)h1 << 16);
    hp.y = (unsigned)h2 | ((unsigned)h3 << 16);
    lp.x = (unsigned)l0 | ((unsigned)l1 << 16);
    lp.y = (unsigned)l2 | ((unsigned)l3 << 16);
    const int rr  = (r < EB) ? r : r - EB;
    const int off = rr * SA + c4 * 4;
    ushort_t *dH = (r < EB) ? sXsH : sXdH;
    ushort_t *dL = (r < EB) ? sXsL : sXdL;
    *(uint2 *)(dH + off) = hp;
    *(uint2 *)(dL + off) = lp;
  }
  // Stage edge_attr tile (same split)
  for (int i = t; i < EB * 16; i += 256) {       // 32 rows x 16 float4
    const int r  = i >> 4;
    const int c4 = i & 15;
    int e = eb0 + r;
    if (e > E - 1) e = E - 1;
    float4 v = ((const float4 *)(ea + (size_t)e * EDGE_DIM))[c4];
    ushort_t h0, l0, h1, l1, h2, l2, h3, l3;
    split_bf(v.x, h0, l0); split_bf(v.y, h1, l1);
    split_bf(v.z, h2, l2); split_bf(v.w, h3, l3);
    uint2 hp, lp;
    hp.x = (unsigned)h0 | ((unsigned)h1 << 16);
    hp.y = (unsigned)h2 | ((unsigned)h3 << 16);
    lp.x = (unsigned)l0 | ((unsigned)l1 << 16);
    lp.y = (unsigned)l2 | ((unsigned)l3 << 16);
    const int off = r * SE + c4 * 4;
    *(uint2 *)(sEAH + off) = hp;
    *(uint2 *)(sEAL + off) = lp;
  }
  __syncthreads();

  const int lane = t & 31;
  const int wave = t >> 5;               // = output N-tile
  const float bn = b_node[wave * 16 + (lane & 15)];
  const float be = b_edge[wave * 16 + (lane & 15)];
  const float bc = b_comb[wave * 16 + (lane & 15)];

  const ushort_t *BNw = BN + (size_t)wave * 8 * 1024;  // 8 K-chunks / ntile
  const ushort_t *BCw = BC + (size_t)wave * 8 * 1024;
  const ushort_t *BEw = BE + (size_t)wave * 2 * 1024;  // 2 K-chunks / ntile
  const ushort_t *BSw = BS + (size_t)wave * 2 * 1024;

  // pe = relu(edge_attr @ W_edge.T + b_edge)   (shared by both directions)
  v8f pe0 = splat8(be), pe1 = splat8(be);
  mm_bf16(pe0, pe1, sEAH, sEAL, SE, BEw, lane, 2);
  relu8(pe0); relu8(pe1);
  store_tile_bf(sPEH, sPEL, pe0, 0, wave, lane);
  store_tile_bf(sPEH, sPEL, pe1, 1, wave, lane);

  // pn forward = relu([x_src | x_dst] @ W_node.T + b_node)
  v8f p0 = splat8(bn), p1 = splat8(bn);
  mm_bf16(p0, p1, sXsH, sXsL, SA, BNw, lane, 4);
  mm_bf16(p0, p1, sXdH, sXdL, SA, BNw + 4 * 1024, lane, 4);
  relu8(p0); relu8(p1);
  store_tile_bf(sPNH, sPNL, p0, 0, wave, lane);
  store_tile_bf(sPNH, sPNL, p1, 1, wave, lane);
  __syncthreads();

  // comb forward = relu([pn_f | pe] @ W_comb.T + b_comb)
  v8f cf0 = splat8(bc), cf1 = splat8(bc);
  mm_bf16(cf0, cf1, sPNH, sPNL, SA, BCw, lane, 4);
  mm_bf16(cf0, cf1, sPEH, sPEL, SA, BCw + 4 * 1024, lane, 4);
  relu8(cf0); relu8(cf1);
  __syncthreads();  // all waves done reading sPN before overwrite

  // pn reverse = relu([x_dst | x_src] @ W_node.T + b_node)
  v8f q0 = splat8(bn), q1 = splat8(bn);
  mm_bf16(q0, q1, sXdH, sXdL, SA, BNw, lane, 4);
  mm_bf16(q0, q1, sXsH, sXsL, SA, BNw + 4 * 1024, lane, 4);
  relu8(q0); relu8(q1);
  store_tile_bf(sPNH, sPNL, q0, 0, wave, lane);
  store_tile_bf(sPNH, sPNL, q1, 1, wave, lane);
  __syncthreads();

  // comb reverse
  v8f cr0 = splat8(bc), cr1 = splat8(bc);
  mm_bf16(cr0, cr1, sPNH, sPNL, SA, BCw, lane, 4);
  mm_bf16(cr0, cr1, sPEH, sPEL, SA, BCw + 4 * 1024, lane, 4);
  relu8(cr0); relu8(cr1);

  // skip = edge_attr @ W_skip.T   (no bias)
  v8f sk0 = splat8(0.f), sk1 = splat8(0.f);
  mm_bf16(sk0, sk1, sEAH, sEAL, SE, BSw, lane, 2);

  // h = 0.5*(comb_f + comb_r) + skip  -> LDS for LayerNorm
#pragma unroll
  for (int r = 0; r < 8; ++r) {
    cf0[r] = (cf0[r] + cr0[r]) * 0.5f + sk0[r];
    cf1[r] = (cf1[r] + cr1[r]) * 0.5f + sk1[r];
  }
  store_tile_f32(sO, cf0, 0, wave, lane);
  store_tile_f32(sO, cf1, 1, wave, lane);
  __syncthreads();

  // ---- LayerNorm over the 128 features, 8 threads per edge-row ----
  {
    const int row = t >> 3;   // 0..31
    const int j   = t & 7;    // 16 cols per thread
    const float *rbuf = sO + row * STRF + j * 16;
    float s = 0.f, ss = 0.f;
#pragma unroll
    for (int c = 0; c < 16; ++c) { float v = rbuf[c]; s += v; ss += v * v; }
    sRs[t] = s; sRss[t] = ss;
    __syncthreads();
    if (j == 0) {
      float S = 0.f, SS = 0.f;
#pragma unroll
      for (int jj = 0; jj < 8; ++jj) {
        S += sRs[(row << 3) + jj];
        SS += sRss[(row << 3) + jj];
      }
      const float mu  = S * (1.0f / OUT_DIM);
      const float var = SS * (1.0f / OUT_DIM) - mu * mu;
      sMU[row]  = mu;
      sRSD[row] = rsqrtf(var + LN_EPS);
    }
    __syncthreads();
    const int e = eb0 + row;
    if (e < E) {
      const float mu = sMU[row], rs = sRSD[row];
      float *og = out + (size_t)e * OUT_DIM + j * 16;
      const float *gb = gamma + j * 16;
      const float *bb = beta + j * 16;
#pragma unroll
      for (int c = 0; c < 16; ++c)
        og[c] = (rbuf[c] - mu) * rs * gb[c] + bb[c];
    }
  }
}

extern "C" void kernel_launch(void *const *d_in, const int *in_sizes, int n_in,
                              void *d_out, int out_size, void *d_ws,
                              size_t ws_size, hipStream_t stream) {
  const float *x         = (const float *)d_in[0];
  const long long *eidx  = (const long long *)d_in[1];
  const float *edge_attr = (const float *)d_in[2];
  const float *W_edge    = (const float *)d_in[3];
  const float *b_edge    = (const float *)d_in[4];
  const float *W_node    = (const float *)d_in[5];
  const float *b_node    = (const float *)d_in[6];
  const float *W_comb    = (const float *)d_in[7];
  const float *b_comb    = (const float *)d_in[8];
  const float *W_skip    = (const float *)d_in[9];
  const float *gamma     = (const float *)d_in[10];
  const float *beta      = (const float *)d_in[11];
  float *out = (float *)d_out;
  const int E = in_sizes[1] / 2;  // edge_index is [2, E]

  // bf16 hi/lo B-fragment workspace: 128KB + 128KB + 32KB + 32KB = 320KB
  ushort_t *BN = (ushort_t *)d_ws;       // W_node (K=256): 8nt * 8kc * 1024
  ushort_t *BC = BN + 8 * 8 * 1024;      // W_comb (K=256)
  ushort_t *BE = BC + 8 * 8 * 1024;      // W_edge (K=64):  8nt * 2kc * 1024
  ushort_t *BS = BE + 8 * 2 * 1024;      // W_skip (K=64)

  pack_wt_bf<<<(8 * 8 * 32 + 255) / 256, 256, 0, stream>>>(W_node, BN, 256);
  pack_wt_bf<<<(8 * 8 * 32 + 255) / 256, 256, 0, stream>>>(W_comb, BC, 256);
  pack_wt_bf<<<(8 * 2 * 32 + 255) / 256, 256, 0, stream>>>(W_edge, BE, 64);
  pack_wt_bf<<<(8 * 2 * 32 + 255) / 256, 256, 0, stream>>>(W_skip, BS, 64);

  const int blocks = (E + EB - 1) / EB;
  edge_block_wmma<<<blocks, 256, 0, stream>>>(x, eidx, edge_attr, b_edge,
                                              b_node, b_comb, gamma, beta, BN,
                                              BC, BE, BS, out, E);
}
